// NameDecoder_87325275062390
// MI455X (gfx1250) — compile-verified, matrix operand order checked by reading, not run on hardware
//
#include <hip/hip_runtime.h>
#include <hip/hip_bf16.h>
#include <math.h>

// Problem constants (match reference)
#define NV 50257
#define ND 1024
#define NH 1024
#define NL 2
#define NB 64
#define NT 32
#define VPAD 50272   // multiple of 16, padded column count for logits scratch

typedef __attribute__((ext_vector_type(16))) __bf16 v16bf;
typedef __attribute__((ext_vector_type(8)))  float  v8f;
typedef __attribute__((ext_vector_type(4)))  unsigned u32x4;
typedef __attribute__((ext_vector_type(8)))  int      i32x8;
typedef __attribute__((ext_vector_type(4)))  int      i32x4;

// ---------------------------------------------------------------------------
// One-time weight conversion fp32 -> bf16 (keeps w_proj L2-resident: 103 MB)
// ---------------------------------------------------------------------------
__global__ void cvt_f32_bf16_kernel(const float* __restrict__ src,
                                    __hip_bfloat16* __restrict__ dst,
                                    long n) {
    long i = (long)blockIdx.x * blockDim.x + threadIdx.x;
    long stride = (long)gridDim.x * blockDim.x;
    for (; i < n; i += stride) dst[i] = __float2bfloat16(src[i]);
}

__global__ void init_h_kernel(const float* __restrict__ ih,
                              float* __restrict__ hf,
                              __hip_bfloat16* __restrict__ hbf,
                              int n) {
    int i = blockIdx.x * blockDim.x + threadIdx.x;
    if (i < n) {
        float v = ih[i];
        hf[i]  = v;
        hbf[i] = __float2bfloat16(v);
    }
}

__global__ void embed_kernel(const float* __restrict__ emb,
                             const int* __restrict__ curtok,
                             const int* __restrict__ start_tok,
                             int t,
                             __hip_bfloat16* __restrict__ x) {
    int i = blockIdx.x * blockDim.x + threadIdx.x;
    if (i >= NB * ND) return;
    int b = i >> 10;
    int k = i & (ND - 1);
    int tok = (t == 0) ? start_tok[0] : curtok[b];
    x[i] = __float2bfloat16(emb[(long)tok * ND + k]);
}

// ---------------------------------------------------------------------------
// Fused GRU layer (unchanged from round 1: already a good WMMA+vmem mix).
// Grid: NH/16 = 64 blocks; Block: 128 threads = 4 wave32.
// ---------------------------------------------------------------------------
__global__ __launch_bounds__(128)
void gru_layer_kernel(const __hip_bfloat16* __restrict__ x,
                      const __hip_bfloat16* __restrict__ wi,
                      const __hip_bfloat16* __restrict__ wh,
                      const float* __restrict__ bi,
                      const float* __restrict__ bh,
                      const __hip_bfloat16* __restrict__ hbf_in,
                      float* __restrict__ hf,
                      __hip_bfloat16* __restrict__ hbf_out,
                      __hip_bfloat16* __restrict__ xout) {
    const int lane = threadIdx.x & 31;
    const int wv   = threadIdx.x >> 5;
    const int m0   = wv * 16;
    const int j0   = blockIdx.x * 16;
    const int rowA = m0 + (lane & 15);
    const int koff = (lane >> 4) * 16;
    const long colR = (long)(0 * NH + j0 + (lane & 15));
    const long colZ = (long)(1 * NH + j0 + (lane & 15));
    const long colN = (long)(2 * NH + j0 + (lane & 15));

    v8f aIr = {}, aIz = {}, aIn = {}, aHr = {}, aHz = {}, aHn = {};

    for (int k0 = 0; k0 < ND; k0 += 32) {
        v16bf ax  = *(const v16bf*)(&x[(long)rowA * ND + k0 + koff]);
        v16bf ah  = *(const v16bf*)(&hbf_in[(long)rowA * NH + k0 + koff]);
        v16bf bIr = *(const v16bf*)(&wi[colR * ND + k0 + koff]);
        v16bf bIz = *(const v16bf*)(&wi[colZ * ND + k0 + koff]);
        v16bf bIn = *(const v16bf*)(&wi[colN * ND + k0 + koff]);
        v16bf bHr = *(const v16bf*)(&wh[colR * NH + k0 + koff]);
        v16bf bHz = *(const v16bf*)(&wh[colZ * NH + k0 + koff]);
        v16bf bHn = *(const v16bf*)(&wh[colN * NH + k0 + koff]);
        aIr = __builtin_amdgcn_wmma_f32_16x16x32_bf16(false, ax, false, bIr, (short)0, aIr, false, false);
        aIz = __builtin_amdgcn_wmma_f32_16x16x32_bf16(false, ax, false, bIz, (short)0, aIz, false, false);
        aIn = __builtin_amdgcn_wmma_f32_16x16x32_bf16(false, ax, false, bIn, (short)0, aIn, false, false);
        aHr = __builtin_amdgcn_wmma_f32_16x16x32_bf16(false, ah, false, bHr, (short)0, aHr, false, false);
        aHz = __builtin_amdgcn_wmma_f32_16x16x32_bf16(false, ah, false, bHz, (short)0, aHz, false, false);
        aHn = __builtin_amdgcn_wmma_f32_16x16x32_bf16(false, ah, false, bHn, (short)0, aHn, false, false);
    }

    const int n  = lane & 15;
    const int mh = (lane >> 4) * 8;
    const int j  = j0 + n;
    const float biR = bi[j], biZ = bi[NH + j], biN = bi[2 * NH + j];
    const float bhR = bh[j], bhZ = bh[NH + j], bhN = bh[2 * NH + j];

#pragma unroll
    for (int i = 0; i < 8; ++i) {
        const int b = m0 + mh + i;
        const float ir  = aIr[i] + biR;
        const float iz  = aIz[i] + biZ;
        const float inn = aIn[i] + biN;
        const float hr  = aHr[i] + bhR;
        const float hz  = aHz[i] + bhZ;
        const float hn  = aHn[i] + bhN;
        const float r   = 1.0f / (1.0f + __expf(-(ir + hr)));
        const float z   = 1.0f / (1.0f + __expf(-(iz + hz)));
        const float nn  = tanhf(inn + r * hn);
        const float hp  = hf[(long)b * NH + j];
        const float hnew = (1.0f - z) * nn + z * hp;
        hf[(long)b * NH + j] = hnew;
        const __hip_bfloat16 hb = __float2bfloat16(hnew);
        hbf_out[(long)b * NH + j] = hb;
        xout[(long)b * NH + j]    = hb;
    }
}

// ---------------------------------------------------------------------------
// Projection: logits[64, V] = x @ w_proj^T + b_proj.
// Block = 128 threads (4 waves) covers 64 columns x all 64 rows.
// Wave w owns column tile j0+16w with 4 accumulators (4 row tiles), so each
// B (weight) fragment feeds 4 WMMAs and every weight byte is loaded once.
// x is DMA'd into LDS by the Tensor Data Mover (TDM) and A fragments come
// from LDS (ds_load), keeping activations on-WGP while weights stream L2.
// ---------------------------------------------------------------------------
__global__ __launch_bounds__(128)
void proj_kernel(const __hip_bfloat16* __restrict__ x,   // [NB, NH] bf16
                 const __hip_bfloat16* __restrict__ wp,  // [NV, NH] bf16
                 const float* __restrict__ bp,           // [NV]
                 float* __restrict__ logits) {           // [NB, VPAD]
    extern __shared__ __hip_bfloat16 xlds[];             // [NB * NH] = 128 KB

    // --- TDM: tensor_load_to_lds of the whole x tile (2D: 1024 x 64, 2B) ---
    if (threadIdx.x == 0) {
        const unsigned long long ga = (unsigned long long)(uintptr_t)x;
        // D# group0: count=1 | lds_addr=0 (dynamic-LDS base) | global_addr | type=2
        u32x4 g0 = { 1u,
                     0u,
                     (unsigned)(ga & 0xFFFFFFFFu),
                     (unsigned)((ga >> 32) & 0x01FFFFFFu) | (2u << 30) };
        // D# group1: data_size=2B; tensor_dim0=1024; tensor_dim1=64;
        //            tile_dim0=1024; tile_dim1=64; tensor_dim0_stride=1024
        i32x8 g1 = { (int)0x00010000,          // wg_mask=0, data_size=1 (2 bytes)
                     (int)(1024u << 16),       // tensor_dim0[15:0] << 16
                     (int)(64u << 16),         // tensor_dim0 hi=0 | tensor_dim1 lo=64
                     (int)(1024u << 16),       // tensor_dim1 hi=0 | tile_dim0=1024
                     64,                       // tile_dim1=64, tile_dim2=0
                     1024,                     // tensor_dim0_stride[31:0]
                     0, 0 };
        i32x4 gz = { 0, 0, 0, 0 };
#if defined(__clang_major__) && (__clang_major__ >= 23)
        i32x8 gz8 = { 0, 0, 0, 0, 0, 0, 0, 0 };
        __builtin_amdgcn_tensor_load_to_lds(g0, g1, gz, gz, gz8, 0);
#else
        __builtin_amdgcn_tensor_load_to_lds(g0, g1, gz, gz, 0);
#endif
        __builtin_amdgcn_s_wait_tensorcnt(0);
    }
    __syncthreads();

    const int lane = threadIdx.x & 31;
    const int wv   = threadIdx.x >> 5;
    const int j0   = blockIdx.x * 64 + wv * 16;   // this wave's column tile
    const int koff = (lane >> 4) * 16;
    const int rA   = lane & 15;
    int colB = j0 + (lane & 15);
    if (colB >= NV) colB = NV - 1;                // clamp loads on tail tile

    v8f acc0 = {}, acc1 = {}, acc2 = {}, acc3 = {};
    for (int k0 = 0; k0 < NH; k0 += 32) {
        if (k0 + 96 <= NH)
            __builtin_prefetch(&wp[(long)colB * NH + k0 + 64], 0, 1);
        v16bf b  = *(const v16bf*)(&wp[(long)colB * NH + k0 + koff]);
        v16bf a0 = *(const v16bf*)(&xlds[(long)(0  + rA) * NH + k0 + koff]);
        v16bf a1 = *(const v16bf*)(&xlds[(long)(16 + rA) * NH + k0 + koff]);
        v16bf a2 = *(const v16bf*)(&xlds[(long)(32 + rA) * NH + k0 + koff]);
        v16bf a3 = *(const v16bf*)(&xlds[(long)(48 + rA) * NH + k0 + koff]);
        acc0 = __builtin_amdgcn_wmma_f32_16x16x32_bf16(false, a0, false, b, (short)0, acc0, false, false);
        acc1 = __builtin_amdgcn_wmma_f32_16x16x32_bf16(false, a1, false, b, (short)0, acc1, false, false);
        acc2 = __builtin_amdgcn_wmma_f32_16x16x32_bf16(false, a2, false, b, (short)0, acc2, false, false);
        acc3 = __builtin_amdgcn_wmma_f32_16x16x32_bf16(false, a3, false, b, (short)0, acc3, false, false);
    }

    const int n  = lane & 15;
    const int mh = (lane >> 4) * 8;
    const int j  = j0 + n;
    if (j < NV) {
        const float bj = bp[j];
#pragma unroll
        for (int i = 0; i < 8; ++i) {
            logits[(long)(0  + mh + i) * VPAD + j] = acc0[i] + bj;
            logits[(long)(16 + mh + i) * VPAD + j] = acc1[i] + bj;
            logits[(long)(32 + mh + i) * VPAD + j] = acc2[i] + bj;
            logits[(long)(48 + mh + i) * VPAD + j] = acc3[i] + bj;
        }
    }
}

// ---------------------------------------------------------------------------
// Per-batch-row argmax (first-occurrence ties) + log-sum-exp. Grid: NB blocks.
// ---------------------------------------------------------------------------
__global__ __launch_bounds__(256)
void reduce_kernel(const float* __restrict__ logits,
                   int* __restrict__ tok,
                   float* __restrict__ stats) {
    __shared__ float smax[256];
    __shared__ int   sidx[256];
    __shared__ float ssum[256];
    const int b = blockIdx.x;
    const int tid = threadIdx.x;
    const float* row = logits + (long)b * VPAD;

    float mv = -INFINITY;
    int   mi = 0x7fffffff;
    for (int v = tid; v < NV; v += 256) {
        float val = row[v];
        if (val > mv) { mv = val; mi = v; }
    }
    smax[tid] = mv; sidx[tid] = mi;
    __syncthreads();
    for (int s = 128; s > 0; s >>= 1) {
        if (tid < s) {
            float ov = smax[tid + s]; int oi = sidx[tid + s];
            if (ov > smax[tid] || (ov == smax[tid] && oi < sidx[tid])) {
                smax[tid] = ov; sidx[tid] = oi;
            }
        }
        __syncthreads();
    }
    const float rowmax = smax[0];

    float sum = 0.0f;
    for (int v = tid; v < NV; v += 256) sum += __expf(row[v] - rowmax);
    ssum[tid] = sum;
    __syncthreads();
    for (int s = 128; s > 0; s >>= 1) {
        if (tid < s) ssum[tid] += ssum[tid + s];
        __syncthreads();
    }
    if (tid == 0) {
        tok[b]           = sidx[0];
        stats[b * 2]     = rowmax;
        stats[b * 2 + 1] = rowmax + __logf(ssum[0]);
    }
}

__global__ void writeout_kernel(const float* __restrict__ logits,
                                const float* __restrict__ stats,
                                const int* __restrict__ tok,
                                float* __restrict__ out,
                                int t) {
    long i = (long)blockIdx.x * blockDim.x + threadIdx.x;
    const long total = (long)NB * NV;
    if (i < total) {
        const int b = (int)(i / NV);
        const int v = (int)(i % NV);
        out[(long)b * NT * NV + (long)t * NV + v] =
            logits[(long)b * VPAD + v] - stats[b * 2 + 1];
    }
    if (i < NB) {
        out[(long)NB * NT * NV + i * NT + t] = (float)tok[i];
    }
}

// ---------------------------------------------------------------------------
// Host orchestration (graph-capture safe: all launches on `stream`).
// ---------------------------------------------------------------------------
extern "C" void kernel_launch(void* const* d_in, const int* in_sizes, int n_in,
                              void* d_out, int out_size, void* d_ws, size_t ws_size,
                              hipStream_t stream) {
    const float* emb     = (const float*)d_in[0];
    const float* w_ih    = (const float*)d_in[1];
    const float* w_hh    = (const float*)d_in[2];
    const float* b_ih    = (const float*)d_in[3];
    const float* b_hh    = (const float*)d_in[4];
    const float* w_proj  = (const float*)d_in[5];
    const float* b_proj  = (const float*)d_in[6];
    const float* init_h  = (const float*)d_in[7];
    const int*   start_t = (const int*)d_in[9];
    float* out = (float*)d_out;

    char* ws = (char*)d_ws;
    auto carve = [&](size_t bytes) -> char* {
        char* p = ws;
        ws += (bytes + 255) & ~(size_t)255;
        return p;
    };
    __hip_bfloat16* wproj_bf = (__hip_bfloat16*)carve((size_t)NV * NH * 2);
    __hip_bfloat16* wih_bf   = (__hip_bfloat16*)carve((size_t)NL * 3 * NH * ND * 2);
    __hip_bfloat16* whh_bf   = (__hip_bfloat16*)carve((size_t)NL * 3 * NH * NH * 2);
    float*          hf       = (float*)         carve((size_t)NL * NB * NH * 4);
    __hip_bfloat16* hbfA     = (__hip_bfloat16*)carve((size_t)NL * NB * NH * 2);
    __hip_bfloat16* hbfB     = (__hip_bfloat16*)carve((size_t)NL * NB * NH * 2);
    __hip_bfloat16* xA       = (__hip_bfloat16*)carve((size_t)NB * ND * 2);
    __hip_bfloat16* xB       = (__hip_bfloat16*)carve((size_t)NB * NH * 2);
    __hip_bfloat16* xC       = (__hip_bfloat16*)carve((size_t)NB * NH * 2);
    float*          logits   = (float*)         carve((size_t)NB * VPAD * 4);
    float*          stats    = (float*)         carve((size_t)NB * 2 * 4);
    int*            tok      = (int*)           carve((size_t)NB * 4);

    cvt_f32_bf16_kernel<<<4096, 256, 0, stream>>>(w_proj, wproj_bf, (long)NV * NH);
    cvt_f32_bf16_kernel<<<2048, 256, 0, stream>>>(w_ih,   wih_bf,   (long)NL * 3 * NH * ND);
    cvt_f32_bf16_kernel<<<2048, 256, 0, stream>>>(w_hh,   whh_bf,   (long)NL * 3 * NH * NH);
    init_h_kernel<<<(NL * NB * NH + 255) / 256, 256, 0, stream>>>(init_h, hf, hbfA, NL * NB * NH);

    __hip_bfloat16* hb_cur = hbfA;
    __hip_bfloat16* hb_nxt = hbfB;

    const int projBlocks = (NV + 63) / 64;                       // 786
    const int woBlocks   = (int)(((long)NB * NV + 255) / 256);
    const size_t projLds = (size_t)NB * NH * sizeof(__hip_bfloat16);  // 128 KB

    for (int t = 0; t < NT; ++t) {
        embed_kernel<<<(NB * ND + 255) / 256, 256, 0, stream>>>(emb, tok, start_t, t, xA);

        gru_layer_kernel<<<NH / 16, 128, 0, stream>>>(
            xA, wih_bf, whh_bf, b_ih, b_hh,
            hb_cur, hf, hb_nxt, xB);

        gru_layer_kernel<<<NH / 16, 128, 0, stream>>>(
            xB,
            wih_bf + (size_t)3 * NH * ND,
            whh_bf + (size_t)3 * NH * NH,
            b_ih + 3 * NH, b_hh + 3 * NH,
            hb_cur + (size_t)NB * NH,
            hf + (size_t)NB * NH,
            hb_nxt + (size_t)NB * NH,
            xC);

        proj_kernel<<<projBlocks, 128, projLds, stream>>>(xC, wproj_bf, b_proj, logits);
        reduce_kernel<<<NB, 256, 0, stream>>>(logits, tok, stats);
        writeout_kernel<<<woBlocks, 256, 0, stream>>>(logits, stats, tok, out, t);

        __hip_bfloat16* tmp = hb_cur; hb_cur = hb_nxt; hb_nxt = tmp;
    }

    (void)in_sizes; (void)n_in; (void)out_size; (void)ws_size;
}
